// MHPermutMix_56676388438047
// MI455X (gfx1250) — matrix operand order repaired and checked by reading, not compiled, not run on hardware
//
#include <hip/hip_runtime.h>
#include <hip/hip_bf16.h>
#include <math.h>

// ---------------- problem constants ----------------
#define BATCH    256
#define TOKENS   196
#define EMBED    64
#define DFLAT    (TOKENS * EMBED)      // 12544
#define CONCAT   512                   // GEMM K
#define OUTCH    256                   // GEMM N
#define ROWS     (BATCH * TOKENS)      // 50176 GEMM M
#define MTILE    32                    // rows per block
#define NBLOCKS  (ROWS / MTILE)        // 1568
#define YSTR     520                   // padded f16 row stride (512 + 8)
#define HSTR     260                   // padded f32 row stride (256 + 4)
#define LN_EPS   1e-5f

typedef _Float16 v16h __attribute__((ext_vector_type(16)));
typedef _Float16 v8h  __attribute__((ext_vector_type(8)));
typedef _Float16 v4h  __attribute__((ext_vector_type(4)));
typedef float    v8f  __attribute__((ext_vector_type(8)));
typedef float    v4f  __attribute__((ext_vector_type(4)));
typedef int      v4i  __attribute__((ext_vector_type(4)));

// ---------------- W f32 -> f16 conversion (runs every launch; deterministic) ----
__global__ __launch_bounds__(256) void cvt_w_f16(const float* __restrict__ W,
                                                 _Float16* __restrict__ Wh) {
    int i = blockIdx.x * 256 + threadIdx.x;      // 0 .. 131071
    Wh[i] = (_Float16)W[i];
}

// ---------------- fused gather + GEMM + LN/GELU + pool ----------------
__global__ __launch_bounds__(256) void fused_mix(
    const float* __restrict__ xf,        // [B, D]
    const int*   __restrict__ perms,     // [H*D] == [196*512]
    const float* __restrict__ signs,     // [H*D]
    const _Float16* __restrict__ Wh,     // [256, 512] f16 (row-major, K contiguous)
    const float* __restrict__ bias,      // [256]
    const float* __restrict__ gamma,     // [256]
    const float* __restrict__ beta,      // [256]
    float* __restrict__ out)             // [ROWS, 256]
{
    extern __shared__ char smem[];
    _Float16* Y   = (_Float16*)smem;                                   // [32][YSTR] f16
    float*    Hs  = (float*)(smem + MTILE * YSTR * sizeof(_Float16));  // [32][HSTR] f32
    float*    Red = (float*)(smem + MTILE * YSTR * sizeof(_Float16)
                                  + MTILE * HSTR * sizeof(float));     // 576 floats

    const int  tid  = threadIdx.x;
    const long row0 = (long)blockIdx.x * MTILE;

    // ---- stage 1: gather + sign, f32 -> f16 into LDS tile ----
    {
        const int  r    = tid >> 3;            // local row 0..31
        const int  c0   = (tid & 7) * 64;      // 64 consecutive channels
        const long grow = row0 + r;
        const int  b    = (int)(grow / TOKENS);
        const int  n    = (int)(grow % TOKENS);
        const float* xb = xf + (long)b * DFLAT;
        const int flatbase = n * CONCAT;
        _Float16* yrow = Y + r * YSTR;
        #pragma unroll 4
        for (int c = c0; c < c0 + 64; c += 4) {
            const int flat = flatbase + c;
            v4i idx = *(const v4i*)(perms + flat);
            v4f sg  = *(const v4f*)(signs + flat);
            v4h h;
            h.x = (_Float16)(xb[idx.x] * sg.x);
            h.y = (_Float16)(xb[idx.y] * sg.y);
            h.z = (_Float16)(xb[idx.z] * sg.z);
            h.w = (_Float16)(xb[idx.w] * sg.w);
            *(v4h*)(yrow + c) = h;
        }
    }
    __syncthreads();

    // ---- stage 2: WMMA GEMM  h = y @ W^T ----
    const int lane = tid & 31;
    const int wave = tid >> 5;     // 0..7
    const int l15  = lane & 15;
    const int hi   = lane >> 4;    // 0 or 1 (lane half)
    const int ct0  = wave * 2;     // two 16-col tiles per wave

    v8f acc[2][2] = {};            // [row-tile][col-tile]

    for (int k0 = 0; k0 < CONCAT; k0 += 32) {
        // A fragments: 16x32 f16, contiguous K order (matches B):
        // lanes 0-15 row=l, K = k0..k0+15; lanes 16-31: K = k0+16..k0+31
        v16h a[2];
        #pragma unroll
        for (int rt = 0; rt < 2; ++rt) {
            const _Float16* yp = Y + (rt * 16 + l15) * YSTR + k0 + hi * 16;
            v8h lo = *(const v8h*)(yp);
            v8h hh = *(const v8h*)(yp + 8);
            a[rt] = __builtin_shufflevector(lo, hh,
                     0,1,2,3,4,5,6,7, 8,9,10,11,12,13,14,15);
        }
        // B fragments: 32x16 f16; lane<16: col=l15, K=k0..k0+15 contiguous;
        // lane>=16: K=k0+16..k0+31  -> single 32B global load from L2-resident Wh
        v16h bf[2];
        #pragma unroll
        for (int ci = 0; ci < 2; ++ci) {
            const int col = (ct0 + ci) * 16 + l15;
            const _Float16* wp = Wh + (long)col * CONCAT + k0 + hi * 16;
            // Unconditional speculative WGP-scope prefetch of the next K slice
            // (past-the-end addresses are silently dropped; locality 3 -> near scope)
            __builtin_prefetch(wp + 32, 0, 3);
            bf[ci] = *(const v16h*)wp;
        }
        #pragma unroll
        for (int rt = 0; rt < 2; ++rt)
            #pragma unroll
            for (int ci = 0; ci < 2; ++ci)
                acc[rt][ci] = __builtin_amdgcn_wmma_f32_16x16x32_f16(
                    false, a[rt], false, bf[ci],
                    (short)0, acc[rt][ci], false, false);
    }

    // ---- stage 3: accumulators + bias -> LDS (C/D layout: lane half selects M+8) ----
    #pragma unroll
    for (int rt = 0; rt < 2; ++rt) {
        #pragma unroll
        for (int ci = 0; ci < 2; ++ci) {
            const int col = (ct0 + ci) * 16 + l15;
            const float bb = bias[col];
            #pragma unroll
            for (int j = 0; j < 8; ++j) {
                const int r = rt * 16 + hi * 8 + j;
                Hs[r * HSTR + col] = acc[rt][ci][j] + bb;
            }
        }
    }
    __syncthreads();

    // ---- stage 4: LayerNorm stats (two-stage LDS reduction) ----
    {
        const int r = tid >> 3, seg = tid & 7;
        const float* hp = Hs + r * HSTR + seg * 32;
        float s = 0.f, s2 = 0.f;
        #pragma unroll 8
        for (int i = 0; i < 32; ++i) { float v = hp[i]; s += v; s2 += v * v; }
        Red[r * 8 + seg]       = s;
        Red[256 + r * 8 + seg] = s2;
    }
    __syncthreads();
    if (tid < 32) {
        float s = 0.f, s2 = 0.f;
        #pragma unroll
        for (int i = 0; i < 8; ++i) { s += Red[tid * 8 + i]; s2 += Red[256 + tid * 8 + i]; }
        const float mu  = s * (1.0f / OUTCH);
        const float var = s2 * (1.0f / OUTCH) - mu * mu;
        Red[512 + tid] = mu;
        Red[544 + tid] = rsqrtf(var + LN_EPS);
    }
    __syncthreads();

    // ---- stage 5: LN -> exact GELU -> + pair-pooled y -> store ----
    {
        const int r = tid >> 3, seg = tid & 7;
        const float mu = Red[512 + r];
        const float rs = Red[544 + r];
        const _Float16* yrow = Y + r * YSTR;
        float* op = out + (row0 + r) * OUTCH + seg * 32;
        const float* hp = Hs + r * HSTR + seg * 32;
        #pragma unroll 2
        for (int c4 = 0; c4 < 32; c4 += 4) {
            v4f o;
            #pragma unroll
            for (int u = 0; u < 4; ++u) {
                const int c = seg * 32 + c4 + u;
                const float g = (hp[c4 + u] - mu) * rs * gamma[c] + beta[c];
                const float gelu = 0.5f * g * (1.0f + erff(g * 0.70710678118654752f));
                const float pooled = 0.5f * ((float)yrow[2 * c] + (float)yrow[2 * c + 1]);
                o[u] = gelu + pooled;
            }
            *(v4f*)(op + c4) = o;
        }
    }
}

// ---------------- host-side launch ----------------
extern "C" void kernel_launch(void* const* d_in, const int* in_sizes, int n_in,
                              void* d_out, int out_size, void* d_ws, size_t ws_size,
                              hipStream_t stream) {
    const float* x      = (const float*)d_in[0];   // [256,196,64]
    const int*   perms  = (const int*)  d_in[1];   // [8,12544]
    const float* signs  = (const float*)d_in[2];   // [1,8,12544]
    const float* W      = (const float*)d_in[3];   // [256,512]
    const float* bias   = (const float*)d_in[4];   // [256]
    const float* gamma  = (const float*)d_in[5];   // [256]
    const float* beta   = (const float*)d_in[6];   // [256]
    float*       out    = (float*)d_out;           // [50176,256]

    _Float16* Wh = (_Float16*)d_ws;                // needs 256KB of scratch

    // convert W to f16 (rewritten every launch -> deterministic)
    cvt_w_f16<<<(OUTCH * CONCAT) / 256, 256, 0, stream>>>(W, Wh);

    const size_t smem = MTILE * YSTR * sizeof(_Float16)   // y tile (f16)
                      + MTILE * HSTR * sizeof(float)      // h tile (f32)
                      + 576 * sizeof(float);              // reduction scratch
    fused_mix<<<NBLOCKS, 256, smem, stream>>>(x, perms, signs, Wh,
                                              bias, gamma, beta, out);
}
